// GCN_10282151706761
// MI455X (gfx1250) — matrix operand order repaired and checked by reading, compiled, not run on hardware
//
#include <hip/hip_runtime.h>
#include <hip/hip_bf16.h>
#include <math.h>

typedef __attribute__((ext_vector_type(16))) _Float16 v16h;
typedef __attribute__((ext_vector_type(8)))  _Float16 v8h;
typedef __attribute__((ext_vector_type(8)))  float    v8f;

union HV { v16h v; v8h h[2]; };

#define OC 128   // feature width of all hidden layers

// ---------------------------------------------------------------- utilities
__global__ void k_fill_f32(float* p, float v, long n) {
    long i = (long)blockIdx.x * blockDim.x + threadIdx.x;
    if (i < n) p[i] = v;
}

// ---------------------------------------------------------------- degrees
__global__ void k_deg_edges(const int* dst, float* deg, int E) {
    int e = blockIdx.x * blockDim.x + threadIdx.x;
    if (e < E) atomicAdd(&deg[dst[e]], 1.0f);
}

__global__ void k_rsqrt_inplace(float* deg, int N) {
    int i = blockIdx.x * blockDim.x + threadIdx.x;
    if (i < N) deg[i] = rsqrtf(deg[i]);
}

__global__ void k_edge_norm(const int* src, const int* dst, const float* dinv,
                            float* enorm, int E) {
    int e = blockIdx.x * blockDim.x + threadIdx.x;
    if (e < E) enorm[e] = dinv[src[e]] * dinv[dst[e]];
}

// ------------------------------------------------------- embedding + concat
__global__ void k_embed(const int* x,
                        const float* e0, const float* e1, const float* e2,
                        const float* e3, const float* e4, const float* e5,
                        _Float16* xf, int N) {
    long idx = (long)blockIdx.x * blockDim.x + threadIdx.x;
    long total = (long)N * 384;
    if (idx >= total) return;
    int node = (int)(idx / 384);
    int j    = (int)(idx % 384);
    int t = j >> 6;        // which of 6 embedding tables
    int f = j & 63;        // feature within table (IN = 64)
    int row = x[node * 6 + t];
    const float* tab;
    switch (t) {
        case 0: tab = e0; break;  case 1: tab = e1; break;
        case 2: tab = e2; break;  case 3: tab = e3; break;
        case 4: tab = e4; break;  default: tab = e5; break;
    }
    xf[idx] = (_Float16)tab[row * 64 + f];
}

// ----------------------------------------------- B-matrix fragment packing
// Reorder B [K x 128] (f32) into WMMA-fragment order (f16):
//   Bp[(((kt*8 + nt)*32 + lane)*16) + j] = B[(kt*32 + (lane>>4)*16 + j)*128
//                                            + nt*16 + (lane&15)]
// so a B fragment is 32 contiguous bytes per lane (two b128 loads).
__global__ void k_pack_b(const float* B, _Float16* Bp, int K) {
    long idx = (long)blockIdx.x * blockDim.x + threadIdx.x;
    if (idx >= (long)K * OC) return;
    int j    = (int)(idx & 15);
    int lane = (int)((idx >> 4) & 31);
    int nt   = (int)((idx >> 9) & 7);
    int kt   = (int)(idx >> 12);
    int half_id = lane >> 4, c = lane & 15;
    Bp[idx] = (_Float16)B[(size_t)(kt * 32 + half_id * 16 + j) * OC + nt * 16 + c];
}

// --------------------------------------------------------------- WMMA GEMM
// A: [M x K] f16 row-major; Bp: fragment-packed f16; P: [M x 128] f32.
// One wave computes a 16-row x 128-col stripe. All fragment loads are b128.
template <int K>
__global__ void k_gemm_f16_wmma(const _Float16* __restrict__ A,
                                const _Float16* __restrict__ Bp,
                                float* __restrict__ P, int M) {
    int wave = blockIdx.x * (blockDim.x >> 5) + (threadIdx.x >> 5);
    int lane = threadIdx.x & 31;
    int m0 = wave * 16;
    if (m0 >= M) return;          // wave-uniform: EXEC stays all-1s for WMMA

    int half_id = lane >> 4;
    int r = lane & 15;
    int row = m0 + r; if (row >= M) row = M - 1;   // clamp; stores are guarded
    const _Float16* Arow = A + (size_t)row * K;
    const v8h* BpV = (const v8h*)Bp;

    v8f acc[8];
#pragma unroll
    for (int nt = 0; nt < 8; ++nt) acc[nt] = (v8f){0, 0, 0, 0, 0, 0, 0, 0};

#pragma unroll
    for (int kt = 0; kt < K / 32; ++kt) {
        const int k0 = kt * 32;
        HV a;  // lane's A row: [k0+h*16 .. +15] contiguous
        a.h[0] = *(const v8h*)(Arow + k0 + half_id * 16);
        a.h[1] = *(const v8h*)(Arow + k0 + half_id * 16 + 8);
#pragma unroll
        for (int nt = 0; nt < 8; ++nt) {
            long fb = ((long)(kt * 8 + nt) * 32 + lane) * 2;  // in v8h units
            HV b;
            b.h[0] = BpV[fb];
            b.h[1] = BpV[fb + 1];
            acc[nt] = __builtin_amdgcn_wmma_f32_16x16x32_f16(
                false, a.v, false, b.v, (short)0, acc[nt], false, false);
        }
    }
    // C/D layout: lanes 0-15 -> rows v, lanes 16-31 -> rows v+8; col = lane&15
    int c  = lane & 15;
    int mb = half_id * 8;
#pragma unroll
    for (int nt = 0; nt < 8; ++nt) {
#pragma unroll
        for (int v = 0; v < 8; ++v) {
            int orow = m0 + mb + v;
            if (orow < M) P[(size_t)orow * OC + nt * 16 + c] = acc[nt][v];
        }
    }
}

// -------------------------------------------------------- edge aggregation
__global__ void k_scatter(const float* __restrict__ p, const int* __restrict__ src,
                          const int* __restrict__ dst, const float* __restrict__ enorm,
                          float* __restrict__ agg, int E) {
    long idx = (long)blockIdx.x * blockDim.x + threadIdx.x;
    if (idx >= (long)E * OC) return;
    int e = (int)(idx >> 7);
    int f = (int)(idx & (OC - 1));
    int s = src[e], d = dst[e];
    atomicAdd(&agg[(size_t)d * OC + f], p[(size_t)s * OC + f] * enorm[e]);
}

// self-loop + bias + sigmoid; writes f32 (optional) and f16 copies
__global__ void k_finalize(const float* __restrict__ p, const float* __restrict__ agg,
                           const float* __restrict__ dinv, const float* __restrict__ bias,
                           float* hf, _Float16* hh, int N) {
    long idx = (long)blockIdx.x * blockDim.x + threadIdx.x;
    if (idx >= (long)N * OC) return;
    int i = (int)(idx >> 7);
    int f = (int)(idx & (OC - 1));
    float di = dinv[i];
    float v = agg[idx] + p[idx] * di * di + bias[f];
    v = 1.0f / (1.0f + expf(-v));
    if (hf) hf[idx] = v;
    hh[idx] = (_Float16)v;
}

// ------------------------------------------------------------- batch norm
__global__ void k_bn_stats(const float* __restrict__ g, const float* __restrict__ gb1,
                           float* sums, int N) {   // sums[0..127]=sum, [128..255]=sumsq
    int f = threadIdx.x;                           // blockDim.x == 128
    float s = 0.0f, s2 = 0.0f;
    for (int i = blockIdx.x; i < N; i += gridDim.x) {
        float v = g[(size_t)i * OC + f] + gb1[f];
        s += v; s2 += v * v;
    }
    atomicAdd(&sums[f], s);
    atomicAdd(&sums[OC + f], s2);
}

// gate = relu(BN(g)) @ gW2 + gb2 ; one wave per node, 4 features per lane
__global__ void k_gate(const float* __restrict__ g, const float* __restrict__ gb1,
                       const float* __restrict__ sums, const float* __restrict__ gamma,
                       const float* __restrict__ beta, const float* __restrict__ gW2,
                       const float* __restrict__ gb2, float* gate, int N) {
    int wave = blockIdx.x * (blockDim.x >> 5) + (threadIdx.x >> 5);
    int lane = threadIdx.x & 31;
    if (wave >= N) return;
    float invN = 1.0f / (float)N;
    float acc = 0.0f;
#pragma unroll
    for (int q = 0; q < 4; ++q) {
        int f = lane + q * 32;
        float mean = sums[f] * invN;
        float var  = sums[OC + f] * invN - mean * mean;
        float v = g[(size_t)wave * OC + f] + gb1[f];
        v = gamma[f] * (v - mean) * rsqrtf(var + 1e-5f) + beta[f];
        v = fmaxf(v, 0.0f);
        acc += v * gW2[f];
    }
#pragma unroll
    for (int o = 16; o > 0; o >>= 1) acc += __shfl_down(acc, o, 32);
    if (lane == 0) gate[wave] = acc + gb2[0];
}

// --------------------------------------------------------- segment softmax
__device__ __forceinline__ void atomicMaxF(float* addr, float val) {
    int* p = (int*)addr;
    int old = __float_as_int(*addr);
    while (__int_as_float(old) < val) {
        int assumed = old;
        old = atomicCAS(p, assumed, __float_as_int(val));
        if (old == assumed) break;
    }
}

__global__ void k_seg_max(const float* gate, const int* batch, float* m, int N) {
    int i = blockIdx.x * blockDim.x + threadIdx.x;
    if (i < N) atomicMaxF(&m[batch[i]], gate[i]);
}

__global__ void k_seg_exp_sum(const float* gate, const int* batch, const float* m,
                              float* e, float* z, int N) {
    int i = blockIdx.x * blockDim.x + threadIdx.x;
    if (i >= N) return;
    float ex = expf(gate[i] - m[batch[i]]);
    e[i] = ex;
    atomicAdd(&z[batch[i]], ex);
}

__global__ void k_pool(const float* __restrict__ h2, const float* __restrict__ e,
                       const float* __restrict__ z, const int* __restrict__ batch,
                       float* pooled, int N) {
    long idx = (long)blockIdx.x * blockDim.x + threadIdx.x;
    if (idx >= (long)N * OC) return;
    int i = (int)(idx >> 7);
    int f = (int)(idx & (OC - 1));
    int b = batch[i];
    float a = e[i] / z[b];
    atomicAdd(&pooled[(size_t)b * OC + f], h2[idx] * a);
}

// out[g] = sigmoid(pooled[g] . oW + ob); one wave per graph
__global__ void k_out(const float* __restrict__ pooled, const float* __restrict__ oW,
                      const float* __restrict__ ob, float* out, int G) {
    int wave = blockIdx.x * (blockDim.x >> 5) + (threadIdx.x >> 5);
    int lane = threadIdx.x & 31;
    if (wave >= G) return;
    float acc = 0.0f;
#pragma unroll
    for (int q = 0; q < 4; ++q) {
        int f = lane + q * 32;
        acc += pooled[(size_t)wave * OC + f] * oW[f];
    }
#pragma unroll
    for (int o = 16; o > 0; o >>= 1) acc += __shfl_down(acc, o, 32);
    if (lane == 0) out[wave] = 1.0f / (1.0f + expf(-(acc + ob[0])));
}

// ------------------------------------------------------------------ driver
static inline size_t alignup(size_t x) { return (x + 255) & ~(size_t)255; }
static inline int gblk(long n, int t) { return (int)((n + t - 1) / t); }

extern "C" void kernel_launch(void* const* d_in, const int* in_sizes, int n_in,
                              void* d_out, int out_size, void* d_ws, size_t ws_size,
                              hipStream_t stream) {
    const int N = in_sizes[0] / 6;
    const int E = in_sizes[1] / 2;
    const int G = out_size;                       // [G,1] output

    const int*   x     = (const int*)d_in[0];
    const int*   src   = (const int*)d_in[1];
    const int*   dst   = src + E;
    const int*   batch = (const int*)d_in[2];
    const float* e0 = (const float*)d_in[4];
    const float* e1 = (const float*)d_in[5];
    const float* e2 = (const float*)d_in[6];
    const float* e3 = (const float*)d_in[7];
    const float* e4 = (const float*)d_in[8];
    const float* e5 = (const float*)d_in[9];
    const float* W1 = (const float*)d_in[10];  const float* b1  = (const float*)d_in[11];
    const float* W2 = (const float*)d_in[12];  const float* b2  = (const float*)d_in[13];
    const float* gW1= (const float*)d_in[14];  const float* gb1 = (const float*)d_in[15];
    const float* gamma=(const float*)d_in[16]; const float* beta= (const float*)d_in[17];
    const float* gW2= (const float*)d_in[18];  const float* gb2 = (const float*)d_in[19];
    const float* oW = (const float*)d_in[20];  const float* ob  = (const float*)d_in[21];
    float* out = (float*)d_out;

    // workspace carve-up
    char* w = (char*)d_ws;  size_t off = 0;
    auto take = [&](size_t bytes) { char* p = w + off; off = alignup(off + bytes); return p; };
    _Float16* xfh  = (_Float16*)take((size_t)N * 384 * sizeof(_Float16));
    _Float16* W1p  = (_Float16*)take((size_t)384 * OC * sizeof(_Float16));
    _Float16* W2p  = (_Float16*)take((size_t)OC * OC * sizeof(_Float16));
    _Float16* gW1p = (_Float16*)take((size_t)OC * OC * sizeof(_Float16));
    float*    p    = (float*)   take((size_t)N * OC * sizeof(float));
    float*    agg  = (float*)   take((size_t)N * OC * sizeof(float));
    _Float16* h1h  = (_Float16*)take((size_t)N * OC * sizeof(_Float16));
    float*    h2f  = (float*)   take((size_t)N * OC * sizeof(float));
    _Float16* h2h  = (_Float16*)take((size_t)N * OC * sizeof(_Float16));
    float*    dinv = (float*)   take((size_t)N * sizeof(float));
    float*    enorm= (float*)   take((size_t)E * sizeof(float));
    float*    gate = (float*)   take((size_t)N * sizeof(float));
    float*    ebuf = (float*)   take((size_t)N * sizeof(float));
    float*    sums = (float*)   take((size_t)2 * OC * sizeof(float));
    float*    mbuf = (float*)   take((size_t)G * sizeof(float));
    float*    zbuf = (float*)   take((size_t)G * sizeof(float));
    float*    pooled=(float*)   take((size_t)G * OC * sizeof(float));
    (void)ws_size;

    const int T = 256;
    const long NE = (long)E * OC, NF = (long)N * OC, NX = (long)N * 384;

    // degrees: deg = 1 (self-loop) + edge in-degree; dinv = rsqrt(deg)
    k_fill_f32<<<gblk(N, T), T, 0, stream>>>(dinv, 1.0f, N);
    k_deg_edges<<<gblk(E, T), T, 0, stream>>>(dst, dinv, E);
    k_rsqrt_inplace<<<gblk(N, T), T, 0, stream>>>(dinv, N);
    k_edge_norm<<<gblk(E, T), T, 0, stream>>>(src, dst, dinv, enorm, E);

    // embedding lookup -> xf (f16); weights -> packed f16 fragments
    k_embed<<<gblk(NX, T), T, 0, stream>>>(x, e0, e1, e2, e3, e4, e5, xfh, N);
    k_pack_b<<<gblk(384 * OC, T), T, 0, stream>>>(W1, W1p, 384);
    k_pack_b<<<gblk(OC * OC, T), T, 0, stream>>>(W2, W2p, OC);
    k_pack_b<<<gblk(OC * OC, T), T, 0, stream>>>(gW1, gW1p, OC);

    const int rowTiles = (N + 15) / 16;          // waves needed
    const int gemmBlk  = (rowTiles + 3) / 4;     // 4 waves per 128-thread block

    // layer 1: p = xf @ W1 ; aggregate ; h1 = sigmoid(agg + self + b1)
    k_gemm_f16_wmma<384><<<gemmBlk, 128, 0, stream>>>(xfh, W1p, p, N);
    k_fill_f32<<<gblk(NF, T), T, 0, stream>>>(agg, 0.0f, NF);
    k_scatter<<<gblk(NE, T), T, 0, stream>>>(p, src, dst, enorm, agg, E);
    k_finalize<<<gblk(NF, T), T, 0, stream>>>(p, agg, dinv, b1, nullptr, h1h, N);

    // layer 2
    k_gemm_f16_wmma<OC><<<gemmBlk, 128, 0, stream>>>(h1h, W2p, p, N);
    k_fill_f32<<<gblk(NF, T), T, 0, stream>>>(agg, 0.0f, NF);
    k_scatter<<<gblk(NE, T), T, 0, stream>>>(p, src, dst, enorm, agg, E);
    k_finalize<<<gblk(NF, T), T, 0, stream>>>(p, agg, dinv, b2, h2f, h2h, N);

    // gate head: g = h2 @ gW1 (+gb1 later); BN stats; gate values
    k_gemm_f16_wmma<OC><<<gemmBlk, 128, 0, stream>>>(h2h, gW1p, p, N);
    k_fill_f32<<<gblk(2 * OC, T), T, 0, stream>>>(sums, 0.0f, 2 * OC);
    k_bn_stats<<<1024, 128, 0, stream>>>(p, gb1, sums, N);
    k_gate<<<gblk((long)N * 32, 256), 256, 0, stream>>>(p, gb1, sums, gamma, beta,
                                                        gW2, gb2, gate, N);

    // segment softmax + gated pooling
    k_fill_f32<<<gblk(G, T), T, 0, stream>>>(mbuf, -3.4e38f, G);
    k_fill_f32<<<gblk(G, T), T, 0, stream>>>(zbuf, 0.0f, G);
    k_fill_f32<<<gblk((long)G * OC, T), T, 0, stream>>>(pooled, 0.0f, (long)G * OC);
    k_seg_max<<<gblk(N, T), T, 0, stream>>>(gate, batch, mbuf, N);
    k_seg_exp_sum<<<gblk(N, T), T, 0, stream>>>(gate, batch, mbuf, ebuf, zbuf, N);
    k_pool<<<gblk(NF, T), T, 0, stream>>>(h2f, ebuf, zbuf, batch, pooled, N);

    // readout
    k_out<<<gblk((long)G * 32, 256), 256, 0, stream>>>(pooled, oW, ob, out, G);
}
